// MoE_39436389712592
// MI455X (gfx1250) — matrix operand order repaired and checked by reading, not compiled, not run on hardware
//
#include <hip/hip_runtime.h>
#include <cstdint>
#include <cstddef>

// ---------------- problem constants ----------------
#define T_TOK 4096   // 4 * 1024 tokens
#define DIM   2048
#define HID   1024
#define NEXP  8

typedef __attribute__((ext_vector_type(8)))  float   v8f;
typedef __attribute__((ext_vector_type(16))) __bf16  v16bf;
typedef __attribute__((ext_vector_type(8)))  __bf16  v8bf;

__device__ __forceinline__ int imin(int a, int b) { return a < b ? a : b; }

// load 8 consecutive fp32, scale, convert to bf16 (native cvt), store 16B
__device__ __forceinline__ void stage8(__bf16* dst, const float* src, float s) {
  float4 a = *(const float4*)(src);
  float4 b = *(const float4*)(src + 4);
  v8bf p;
  p[0] = (__bf16)(a.x * s); p[1] = (__bf16)(a.y * s);
  p[2] = (__bf16)(a.z * s); p[3] = (__bf16)(a.w * s);
  p[4] = (__bf16)(b.x * s); p[5] = (__bf16)(b.y * s);
  p[6] = (__bf16)(b.z * s); p[7] = (__bf16)(b.w * s);
  *(v8bf*)dst = p;
}

// A fragment (16x32 bf16): lane holds row=lane&15; contiguous k-chunk [16*kh, 16*kh+16)
__device__ __forceinline__ v16bf load_fragA(const __bf16* rowbase, int khalf) {
  v8bf lo = *(const v8bf*)(rowbase + khalf * 16);
  v8bf hi = *(const v8bf*)(rowbase + khalf * 16 + 8);
  v16bf u;
#pragma unroll
  for (int i = 0; i < 8; ++i) { u[i] = lo[i]; u[8 + i] = hi[i]; }
  return u;
}

// B fragment (32x16 bf16): lane holds col=lane&15; contiguous k-chunk [16*kh, 16*kh+16)
__device__ __forceinline__ v16bf load_fragB(const __bf16* rowbase, int khalf) {
  return *(const v16bf*)(rowbase + khalf * 16);
}

__device__ __forceinline__ v8f wmma_bf16(v16bf a, v16bf b, v8f c) {
  return __builtin_amdgcn_wmma_f32_16x16x32_bf16(false, a, false, b, (short)0, c, false, false);
}

__device__ __forceinline__ float fast_sigmoid(float v) {
  return __builtin_amdgcn_rcpf(1.f + __expf(-v));   // v_rcp_f32, no IEEE div ladder
}

// ---------------- 1) router: sigmoid top-2 ----------------
__global__ void router_kernel(const float* __restrict__ x, const float* __restrict__ gw,
                              int* __restrict__ sel_idx, float* __restrict__ sel_sc) {
  int wid  = threadIdx.x >> 5;
  int lane = threadIdx.x & 31;
  int t = blockIdx.x * 8 + wid;
  if (t >= T_TOK) return;
  const float* xr = x + (size_t)t * DIM;
  float acc[NEXP] = {};
  for (int d = lane; d < DIM; d += 32) {
    float xv = xr[d];
#pragma unroll
    for (int e = 0; e < NEXP; ++e) acc[e] = fmaf(xv, gw[e * DIM + d], acc[e]);
  }
#pragma unroll
  for (int off = 16; off >= 1; off >>= 1)
#pragma unroll
    for (int e = 0; e < NEXP; ++e) acc[e] += __shfl_xor(acc[e], off, 32);
  if (lane == 0) {
    float sc[NEXP];
#pragma unroll
    for (int e = 0; e < NEXP; ++e) sc[e] = fast_sigmoid(acc[e]);
    int b0 = 0; float v0 = sc[0];
#pragma unroll
    for (int e = 1; e < NEXP; ++e) if (sc[e] > v0) { v0 = sc[e]; b0 = e; }
    int b1 = -1; float v1 = -3.4e38f;
#pragma unroll
    for (int e = 0; e < NEXP; ++e) if (e != b0 && sc[e] > v1) { v1 = sc[e]; b1 = e; }
    sel_idx[t * 2 + 0] = b0; sel_sc[t * 2 + 0] = v0;   // ROUTE_SCALE == 1.0
    sel_idx[t * 2 + 1] = b1; sel_sc[t * 2 + 1] = v1;
  }
}

// ---------------- 2) stable expert-sort (deterministic) ----------------
__global__ void build_kernel(const int* __restrict__ sel_idx, const float* __restrict__ sel_sc,
                             int* __restrict__ ptok, float* __restrict__ psc, int* __restrict__ offs) {
  __shared__ int s_cnt[NEXP], s_off[NEXP + 1];
  int tid = threadIdx.x;
  if (tid < NEXP) {
    int c = 0;
    for (int i = 0; i < 2 * T_TOK; ++i) c += (sel_idx[i] == tid);
    s_cnt[tid] = c;
  }
  __syncthreads();
  if (tid == 0) {
    int run = 0;
    for (int e = 0; e < NEXP; ++e) { s_off[e] = run; run += s_cnt[e]; }
    s_off[NEXP] = run;
    for (int e = 0; e <= NEXP; ++e) offs[e] = s_off[e];
  }
  __syncthreads();
  if (tid < NEXP) {
    int pos = s_off[tid];
    for (int i = 0; i < 2 * T_TOK; ++i)
      if (sel_idx[i] == tid) { ptok[pos] = i; psc[pos] = sel_sc[i]; ++pos; }
  }
}

// ---------------- 3/5) FFN1: h = silu(x@W1^T) * (x@W3^T), bf16 out ----------------
// 256 threads (8 waves as 4x2), block tile 64(M) x 128(N), K staged 64 at a time.
// Wave tile 16(M) x 64(N): 16 wmma per stage between barriers.
template <bool ROUTED>
__global__ void ffn1_kernel(const float* __restrict__ x,
                            const float* __restrict__ w1, const float* __restrict__ w3,
                            const int* __restrict__ ptok, const float* __restrict__ psc,
                            const int* __restrict__ offs, __bf16* __restrict__ Hout) {
  int e = blockIdx.y;
  int row_start, row_end;
  if (ROUTED) {
    row_start = offs[e] + blockIdx.x * 64;
    row_end   = offs[e + 1];
    if (row_start >= row_end) return;
  } else {
    row_start = blockIdx.x * 64;
    row_end   = T_TOK;
  }
  int nb = blockIdx.z * 128;

  __shared__ __bf16 As[64 * 64];          // 8 KB
  __shared__ __bf16 B1s[128 * 64];        // 16 KB
  __shared__ __bf16 B3s[128 * 64];        // 16 KB
  __shared__ int    s_tok[64];
  __shared__ float  s_scl[64];

  int tid = threadIdx.x;
  if (tid < 64) {
    if (ROUTED) {
      int rr = imin(row_start + tid, row_end - 1);
      s_tok[tid] = ptok[rr] >> 1;
      s_scl[tid] = psc[rr];
    } else {
      s_tok[tid] = row_start + tid;
      s_scl[tid] = 1.f;
    }
  }
  __syncthreads();

  const float* W1 = w1 + (ROUTED ? (size_t)e * HID * DIM : 0);
  const float* W3 = w3 + (ROUTED ? (size_t)e * HID * DIM : 0);

  // staging maps
  int ar = tid >> 2, ak = (tid & 3) * 8;    // A: 64 rows x (8-chunk at ak, ak+32)
  int br = tid >> 1, bk = (tid & 1) * 8;    // B: 128 rows x (8-chunks at bk+{0,16,32,48})
  // compute maps
  int wid = tid >> 5, lane = tid & 31;
  int wm = wid & 3, wn = wid >> 2;          // 4x2 waves; wave tile 16 x 64
  int lrow = lane & 15, khalf = lane >> 4;

  const float* arow  = x  + (size_t)s_tok[ar] * DIM + ak;
  float ascl = s_scl[ar];
  const float* b1row = W1 + (size_t)(nb + br) * DIM + bk;
  const float* b3row = W3 + (size_t)(nb + br) * DIM + bk;

  v8f acc1[4] = {}; v8f acc3[4] = {};

  for (int kb = 0; kb < DIM; kb += 64) {
    stage8(&As[ar * 64 + ak],      arow + kb,      ascl);
    stage8(&As[ar * 64 + ak + 32], arow + kb + 32, ascl);
#pragma unroll
    for (int c = 0; c < 4; ++c) {
      stage8(&B1s[br * 64 + bk + c * 16], b1row + kb + c * 16, 1.f);
      stage8(&B3s[br * 64 + bk + c * 16], b3row + kb + c * 16, 1.f);
    }
    __syncthreads();

#pragma unroll
    for (int k2 = 0; k2 < 2; ++k2) {     // two 32-wide k sub-blocks
      v16bf a = load_fragA(&As[(wm * 16 + lrow) * 64 + k2 * 32], khalf);
#pragma unroll
      for (int in = 0; in < 4; ++in) {
        const __bf16* bbase1 = &B1s[(wn * 64 + in * 16 + lrow) * 64 + k2 * 32];
        const __bf16* bbase3 = &B3s[(wn * 64 + in * 16 + lrow) * 64 + k2 * 32];
        acc1[in] = wmma_bf16(a, load_fragB(bbase1, khalf), acc1[in]);
        acc3[in] = wmma_bf16(a, load_fragB(bbase3, khalf), acc3[in]);
      }
    }
    __syncthreads();
  }

#pragma unroll
  for (int in = 0; in < 4; ++in)
#pragma unroll
    for (int v = 0; v < 8; ++v) {
      int r = row_start + wm * 16 + v + khalf * 8;
      if (r < row_end) {
        float h1 = acc1[in][v], h3 = acc3[in][v];
        float h = h1 * fast_sigmoid(h1) * h3;          // silu(h1) * h3
        int col = nb + wn * 64 + in * 16 + lrow;
        Hout[(size_t)r * HID + col] = (__bf16)h;
      }
    }
}

// ---------------- 4/6) FFN2: y = h @ W2^T ----------------
// Same tiling: 64(M) x 128(N), BK=64; 8 wmma per wave per stage.
template <bool ROUTED>
__global__ void ffn2_kernel(const __bf16* __restrict__ Hin, const float* __restrict__ w2,
                            const int* __restrict__ ptok, const int* __restrict__ offs,
                            float* __restrict__ out) {
  int e = blockIdx.y;
  int row_start, row_end;
  if (ROUTED) {
    row_start = offs[e] + blockIdx.x * 64;
    row_end   = offs[e + 1];
    if (row_start >= row_end) return;
  } else {
    row_start = blockIdx.x * 64;
    row_end   = T_TOK;
  }
  int nb = blockIdx.z * 128;  // over DIM

  __shared__ __bf16 As[64 * 64];
  __shared__ __bf16 Bs[128 * 64];
  __shared__ int    s_tok[64];

  int tid = threadIdx.x;
  if (tid < 64) {
    if (ROUTED) {
      int rr = imin(row_start + tid, row_end - 1);
      s_tok[tid] = ptok[rr] >> 1;
    } else {
      s_tok[tid] = row_start + tid;
    }
  }
  __syncthreads();

  const float* W2 = w2 + (ROUTED ? (size_t)e * DIM * HID : 0);

  int ar = tid >> 2, ak = (tid & 3) * 8;
  int br = tid >> 1, bk = (tid & 1) * 8;
  int wid = tid >> 5, lane = tid & 31;
  int wm = wid & 3, wn = wid >> 2;
  int lrow = lane & 15, khalf = lane >> 4;

  int rA = ROUTED ? imin(row_start + ar, row_end - 1) : (row_start + ar);
  const __bf16* arow = Hin + (size_t)rA * HID + ak;
  const float*  brow = W2 + (size_t)(nb + br) * HID + bk;

  v8f acc[4] = {};
  for (int kb = 0; kb < HID; kb += 64) {
    // A already bf16: straight 16B copies
    *(v8bf*)&As[ar * 64 + ak]      = *(const v8bf*)(arow + kb);
    *(v8bf*)&As[ar * 64 + ak + 32] = *(const v8bf*)(arow + kb + 32);
#pragma unroll
    for (int c = 0; c < 4; ++c)
      stage8(&Bs[br * 64 + bk + c * 16], brow + kb + c * 16, 1.f);
    __syncthreads();

#pragma unroll
    for (int k2 = 0; k2 < 2; ++k2) {
      v16bf a = load_fragA(&As[(wm * 16 + lrow) * 64 + k2 * 32], khalf);
#pragma unroll
      for (int in = 0; in < 4; ++in) {
        const __bf16* bbase = &Bs[(wn * 64 + in * 16 + lrow) * 64 + k2 * 32];
        acc[in] = wmma_bf16(a, load_fragB(bbase, khalf), acc[in]);
      }
    }
    __syncthreads();
  }

#pragma unroll
  for (int in = 0; in < 4; ++in)
#pragma unroll
    for (int v = 0; v < 8; ++v) {
      int r = row_start + wm * 16 + v + khalf * 8;
      if (r < row_end) {
        int col = nb + wn * 64 + in * 16 + lrow;
        int tok = s_tok[wm * 16 + v + khalf * 8];
        if (ROUTED) atomicAdd(out + (size_t)tok * DIM + col, acc[in][v]);
        else        out[(size_t)tok * DIM + col] = acc[in][v];
      }
    }
}

// ---------------- workspace layout ----------------
static constexpr size_t OFF_SEL_IDX = 0;                          // int[8192]
static constexpr size_t OFF_SEL_SC  = OFF_SEL_IDX + 32768;        // float[8192]
static constexpr size_t OFF_PTOK    = OFF_SEL_SC  + 32768;        // int[8192]
static constexpr size_t OFF_PSC     = OFF_PTOK    + 32768;        // float[8192]
static constexpr size_t OFF_OFFS    = OFF_PSC     + 32768;        // int[16]
static constexpr size_t OFF_H       = OFF_OFFS    + 256;          // bf16[8192*1024]
static constexpr size_t OFF_HS      = OFF_H + (size_t)2 * T_TOK * HID * 2; // bf16[4096*1024]

extern "C" void kernel_launch(void* const* d_in, const int* in_sizes, int n_in,
                              void* d_out, int out_size, void* d_ws, size_t ws_size,
                              hipStream_t stream) {
  const float* x   = (const float*)d_in[0];
  const float* gw  = (const float*)d_in[1];
  const float* w1  = (const float*)d_in[2];
  const float* w2  = (const float*)d_in[3];
  const float* w3  = (const float*)d_in[4];
  const float* ws1 = (const float*)d_in[5];
  const float* ws2 = (const float*)d_in[6];
  const float* ws3 = (const float*)d_in[7];
  float* out = (float*)d_out;

  char* ws = (char*)d_ws;
  int*    sel_idx = (int*)(ws + OFF_SEL_IDX);
  float*  sel_sc  = (float*)(ws + OFF_SEL_SC);
  int*    ptok    = (int*)(ws + OFF_PTOK);
  float*  psc     = (float*)(ws + OFF_PSC);
  int*    offs    = (int*)(ws + OFF_OFFS);
  __bf16* H       = (__bf16*)(ws + OFF_H);
  __bf16* Hs      = (__bf16*)(ws + OFF_HS);

  // 1) router
  router_kernel<<<dim3(T_TOK / 8), 256, 0, stream>>>(x, gw, sel_idx, sel_sc);
  // 2) stable expert sort
  build_kernel<<<dim3(1), 64, 0, stream>>>(sel_idx, sel_sc, ptok, psc, offs);
  // 3) shared expert FFN1 -> Hs
  ffn1_kernel<false><<<dim3(T_TOK / 64, 1, HID / 128), 256, 0, stream>>>(
      x, ws1, ws3, nullptr, nullptr, nullptr, Hs);
  // 4) shared expert FFN2 -> out (plain stores: also initializes out)
  ffn2_kernel<false><<<dim3(T_TOK / 64, 1, DIM / 128), 256, 0, stream>>>(
      Hs, ws2, nullptr, nullptr, out);
  // 5) routed FFN1 (gather + score scaling) -> H
  ffn1_kernel<true><<<dim3(2 * T_TOK / 64, NEXP, HID / 128), 256, 0, stream>>>(
      x, w1, w3, ptok, psc, offs, H);
  // 6) routed FFN2 -> atomic scatter-add onto out
  ffn2_kernel<true><<<dim3(2 * T_TOK / 64, NEXP, DIM / 128), 256, 0, stream>>>(
      H, w2, ptok, offs, out);
}
